// MultiHeadAttention_23502061043845
// MI455X (gfx1250) — 16-bit and 8-bit matrix operand order repaired and checked by reading, compiled, not run on hardware
//
#include <hip/hip_runtime.h>

#define Bn 4
#define Sn 4096
#define Dn 256
#define Mn (Bn * Sn)

typedef __attribute__((ext_vector_type(16))) __bf16 v16bf;
typedef __attribute__((ext_vector_type(8)))  float  v8f;
typedef __attribute__((ext_vector_type(8)))  int    v8i;
typedef unsigned long long ull;

union ABf16 { v16bf v; __bf16 h[16]; unsigned short u[16]; };
union AI8   { v8i v;   ull q[4]; };

__device__ __forceinline__ unsigned short bf16_bits(float f) {
  union { __bf16 h; unsigned short u; } c;
  c.h = (__bf16)f;
  return c.u;
}

__global__ void w_to_bf16t_kernel(const float* __restrict__ src,
                                  unsigned short* __restrict__ dst) {
  const int kk = threadIdx.x;
  const int nn = blockIdx.x;
  dst[(size_t)nn * 256 + kk] = bf16_bits(src[(size_t)kk * 256 + nn]);
}

template <int MODE>
__global__ void gemm256_bf16_kernel(const float* __restrict__ X,
                                    const unsigned short* __restrict__ Wt,
                                    const float* __restrict__ bias,
                                    void* __restrict__ out) {
  const int lane = threadIdx.x & 31;
  const int wv   = threadIdx.x >> 5;
  const int tile = blockIdx.x * 8 + wv;
  const int tm = tile >> 4;
  const int tn = tile & 15;
  if (tm >= Mn / 16) return;

  const int row  = tm * 16 + (lane & 15);
  const int col  = tn * 16 + (lane & 15);
  const int aoff = (lane >= 16) ? 16 : 0;
  const int boff = (lane >= 16) ? 16 : 0;

  v8f acc = {};
  for (int k0 = 0; k0 < 256; k0 += 32) {
    ABf16 a;
    const float* xr = X + (size_t)row * 256 + k0 + aoff;
#pragma unroll
    for (int e = 0; e < 8; ++e)  a.h[e]     = (__bf16)xr[e];
#pragma unroll
    for (int e = 0; e < 8; ++e)  a.h[8 + e] = (__bf16)xr[8 + e];
    ABf16 bb;
    const unsigned short* wr = Wt + (size_t)col * 256 + k0 + boff;
#pragma unroll
    for (int e = 0; e < 16; ++e) bb.u[e] = wr[e];
    acc = __builtin_amdgcn_wmma_f32_16x16x32_bf16(false, a.v, false, bb.v,
                                                  (short)0, acc, false, false);
  }

  const float bval = bias[col];
#pragma unroll
  for (int r = 0; r < 8; ++r) {
    const int m = tm * 16 + r + ((lane >= 16) ? 8 : 0);
    const float y = acc[r] + bval;
    const size_t idx = (size_t)m * 256 + col;
    if (MODE == 0) {
      ((float*)out)[idx] = y;
    } else if (MODE == 1) {
      ((signed char*)out)[idx] = (signed char)((y > 0.f) - (y < 0.f));
    } else {
      ((unsigned short*)out)[idx] = bf16_bits(y);
    }
  }
}

__global__ void nnz_kernel(const signed char* __restrict__ sq,
                           float* __restrict__ nnzf) {
  const int lane = threadIdx.x & 31;
  const int wv   = threadIdx.x >> 5;
  const int row  = blockIdx.x * 8 + wv;
  if (row >= Mn) return;
  const signed char* p = sq + (size_t)row * 256 + lane * 8;
  int c = 0;
#pragma unroll
  for (int i = 0; i < 8; ++i) c += (p[i] != 0);
#pragma unroll
  for (int off = 16; off > 0; off >>= 1) c += __shfl_xor(c, off, 32);
  if (lane == 0) nnzf[row] = (float)c;
}

__global__ void logits_softmax_kernel(const signed char* __restrict__ s_k,
                                      const signed char* __restrict__ s_q,
                                      const float* __restrict__ nnzf,
                                      const float* __restrict__ mask,
                                      float* __restrict__ attn) {
  extern __shared__ float lds[];
  const int lane = threadIdx.x & 31;
  const int wv   = threadIdx.x >> 5;
  const int kt   = blockIdx.x & 255;
  const int b    = blockIdx.x >> 8;

  const int krow = kt * 16 + (lane & 15);
  const signed char* skrow = s_k + ((size_t)b * Sn + krow) * Dn;
  const int loff = (lane >= 16) ? 8 : 0;
  AI8 aA[4];
#pragma unroll
  for (int st = 0; st < 4; ++st) {
    const int d0 = st * 64 + loff;
    aA[st].q[0] = *(const ull*)(skrow + d0 + 0);
    aA[st].q[1] = *(const ull*)(skrow + d0 + 16);
    aA[st].q[2] = *(const ull*)(skrow + d0 + 32);
    aA[st].q[3] = *(const ull*)(skrow + d0 + 48);
  }
  const int koffB = (lane >= 16) ? 8 : 0;
  for (int qt = wv; qt < Sn / 16; qt += 8) {
    const int qcol = qt * 16 + (lane & 15);
    const signed char* sqrow = s_q + ((size_t)b * Sn + qcol) * Dn;
    v8i acc = {};
#pragma unroll
    for (int st = 0; st < 4; ++st) {
      const int d0 = st * 64 + koffB;
      AI8 bB;
      bB.q[0] = *(const ull*)(sqrow + d0 + 0);
      bB.q[1] = *(const ull*)(sqrow + d0 + 16);
      bB.q[2] = *(const ull*)(sqrow + d0 + 32);
      bB.q[3] = *(const ull*)(sqrow + d0 + 48);
      acc = __builtin_amdgcn_wmma_i32_16x16x64_iu8(true, aA[st].v, true, bB.v,
                                                   acc, false, false);
    }
    const float nq = nnzf[(size_t)b * Sn + qcol];
#pragma unroll
    for (int r = 0; r < 8; ++r) {
      const int kl = r + ((lane >= 16) ? 8 : 0);
      lds[(size_t)kl * Sn + qt * 16 + (lane & 15)] =
          ((float)acc[r] + nq) * (1.0f / 1024.0f);
    }
  }
  __syncthreads();

  for (int rr = 0; rr < 2; ++rr) {
    const int kl = wv + rr * 8;
    float* row = lds + (size_t)kl * Sn;
    float mx = -3.402823466e38f;
    for (int i = lane; i < Sn; i += 32) mx = fmaxf(mx, row[i]);
#pragma unroll
    for (int off = 16; off > 0; off >>= 1) mx = fmaxf(mx, __shfl_xor(mx, off, 32));
    float sum = 0.f;
    for (int i = lane; i < Sn; i += 32) {
      const float e = __expf(row[i] - mx);
      row[i] = e;
      sum += e;
    }
#pragma unroll
    for (int off = 16; off > 0; off >>= 1) sum += __shfl_xor(sum, off, 32);
    const float inv = 1.0f / sum;
    const size_t base = ((size_t)b * Sn + kt * 16 + kl) * Sn;
    const float* mrow = mask + base;
    float* arow = attn + base;
    for (int i = lane; i < Sn; i += 32) arow[i] = row[i] * inv * mrow[i];
  }
}

__global__ void av_tanh_kernel(const float* __restrict__ attn,
                               const unsigned short* __restrict__ vp,
                               const float* __restrict__ anchor,
                               float* __restrict__ mid) {
  __shared__ unsigned short slab[2][32 * 256];
  const int tid  = threadIdx.x;
  const int lane = tid & 31;
  const int wv   = tid >> 5;
  const int qt   = blockIdx.x & 255;
  const int b    = blockIdx.x >> 8;

  const int qrow  = qt * 16 + (lane & 15);
  const int aoff  = (lane >= 16) ? 16 : 0;
  const int d0    = (2 * wv) * 16 + (lane & 15);
  const int d1    = d0 + 16;
  const int koffB = (lane >= 16) ? 16 : 0;

  const float* attnB = attn + (size_t)b * Sn * Sn;
  const unsigned short* vpB = vp + (size_t)b * Sn * Dn;

  auto stage = [&](int buf, int k0) {
    const unsigned lbase =
        (unsigned)(size_t)(const void*)&slab[buf][0] + (unsigned)tid * 16u;
    const ull gbase = (ull)(const void*)((const char*)(vpB + (size_t)k0 * Dn) +
                                         (size_t)tid * 16);
#pragma unroll
    for (int i = 0; i < 4; ++i) {
      const unsigned laddr = lbase + (unsigned)i * 4096u;
      const ull gaddr = gbase + (ull)i * 4096u;
      asm volatile("global_load_async_to_lds_b128 %0, %1, off"
                   :: "v"(laddr), "v"(gaddr) : "memory");
    }
  };

  v8f acc0 = {}, acc1 = {};
  stage(0, 0);
  for (int k0 = 0; k0 < Sn; k0 += 32) {
    const int buf = (k0 >> 5) & 1;
    asm volatile("s_wait_asynccnt 0" ::: "memory");
    __syncthreads();
    if (k0 + 32 < Sn) stage(buf ^ 1, k0 + 32);

    ABf16 a;
#pragma unroll
    for (int e = 0; e < 16; ++e) {
      const int k = k0 + aoff + e;
      a.h[e] = (__bf16)attnB[(size_t)k * Sn + qrow];
    }
    const unsigned short* sl = &slab[buf][(size_t)koffB * Dn];
    ABf16 b0, b1;
#pragma unroll
    for (int e = 0; e < 16; ++e) {
      b0.u[e] = sl[(size_t)e * Dn + d0];
      b1.u[e] = sl[(size_t)e * Dn + d1];
    }
    acc0 = __builtin_amdgcn_wmma_f32_16x16x32_bf16(false, a.v, false, b0.v,
                                                   (short)0, acc0, false, false);
    acc1 = __builtin_amdgcn_wmma_f32_16x16x32_bf16(false, a.v, false, b1.v,
                                                   (short)0, acc1, false, false);
  }

#pragma unroll
  for (int r = 0; r < 8; ++r) {
    const int q = qt * 16 + r + ((lane >= 16) ? 8 : 0);
    const float m0 = tanhf(acc0[r] * anchor[(size_t)q * Dn + d0]);
    const float m1 = tanhf(acc1[r] * anchor[(size_t)q * Dn + d1]);
    mid[((size_t)b * Sn + q) * Dn + d0] = m0;
    mid[((size_t)b * Sn + q) * Dn + d1] = m1;
  }
}

extern "C" void kernel_launch(void* const* d_in, const int* in_sizes, int n_in,
                              void* d_out, int out_size, void* d_ws, size_t ws_size,
                              hipStream_t stream) {
  const float* v       = (const float*)d_in[0];
  const float* k       = (const float*)d_in[1];
  const float* q       = (const float*)d_in[2];
  const float* mask    = (const float*)d_in[3];
  const float* wq_k    = (const float*)d_in[4];
  const float* wq_b    = (const float*)d_in[5];
  const float* wk_k    = (const float*)d_in[6];
  const float* wk_b    = (const float*)d_in[7];
  const float* wv_k    = (const float*)d_in[8];
  const float* wv_b    = (const float*)d_in[9];
  const float* anchor  = (const float*)d_in[10];
  const float* dense_k = (const float*)d_in[11];
  const float* dense_b = (const float*)d_in[12];

  float* out_f  = (float*)d_out;
  float* attn_f = out_f + (size_t)Bn * Sn * Dn;

  char* ws = (char*)d_ws;
  signed char*    s_q   = (signed char*)(ws);
  signed char*    s_k   = (signed char*)(ws + (size_t)(16u << 20));
  unsigned short* vp    = (unsigned short*)(ws + (size_t)(32u << 20));
  float*          nnzf  = (float*)(ws + (size_t)(64u << 20));
  float*          mid   = (float*)(ws + (size_t)(65u << 20));
  unsigned short* wq_t  = (unsigned short*)(ws + (size_t)(81u << 20));
  unsigned short* wk_t  = wq_t + 256 * 256;
  unsigned short* wv_t  = wk_t + 256 * 256;
  unsigned short* wd_t  = wv_t + 256 * 256;

  w_to_bf16t_kernel<<<256, 256, 0, stream>>>(wq_k,    wq_t);
  w_to_bf16t_kernel<<<256, 256, 0, stream>>>(wk_k,    wk_t);
  w_to_bf16t_kernel<<<256, 256, 0, stream>>>(wv_k,    wv_t);
  w_to_bf16t_kernel<<<256, 256, 0, stream>>>(dense_k, wd_t);

  const int gemmBlocks = (Mn / 16) * (Dn / 16) / 8;

  gemm256_bf16_kernel<1><<<gemmBlocks, 256, 0, stream>>>(q, wq_t, wq_b, s_q);
  gemm256_bf16_kernel<1><<<gemmBlocks, 256, 0, stream>>>(k, wk_t, wk_b, s_k);
  gemm256_bf16_kernel<2><<<gemmBlocks, 256, 0, stream>>>(v, wv_t, wv_b, vp);

  nnz_kernel<<<Mn / 8, 256, 0, stream>>>(s_q, nnzf);

  logits_softmax_kernel<<<Bn * (Sn / 16), 256, (size_t)16 * Sn * sizeof(float),
                          stream>>>(s_k, s_q, nnzf, mask, attn_f);

  av_tanh_kernel<<<Bn * (Sn / 16), 256, 0, stream>>>(attn_f, vp, anchor, mid);

  gemm256_bf16_kernel<0><<<gemmBlocks, 256, 0, stream>>>(mid, wd_t, dense_b,
                                                         out_f);
}
